// multi_head_attention_8538394984539
// MI455X (gfx1250) — matrix operand order repaired and checked by reading, not compiled, not run on hardware
//
#include <hip/hip_runtime.h>
#include <hip/hip_bf16.h>

typedef __bf16 bf16_t;
typedef __attribute__((ext_vector_type(16))) __bf16 v16bf;
typedef __attribute__((ext_vector_type(8)))  float  v8f;
typedef __attribute__((ext_vector_type(4)))  unsigned int u32x4;
typedef __attribute__((ext_vector_type(4)))  float  f32x4;

union Frag  { v16bf v; u32x4 q[2]; };
union Pack8 { bf16_t e[8]; u32x4 q; };

#define D_MODEL   1024
#define NUM_HEADS 16
#define HEAD_DIM  64
#define SEQ       2048
#define BATCH     4
#define M_TOT     (BATCH * SEQ)

// ---------------------------------------------------------------------------
// CDNA5 async global->LDS copy (ASYNCcnt-tracked), GV addressing mode.
// ---------------------------------------------------------------------------
__device__ __forceinline__ void async_copy_b128(const bf16_t* lds, const bf16_t* g) {
    const unsigned lds_off = (unsigned)(unsigned long long)lds;
    const unsigned long long ga = (unsigned long long)g;
    asm volatile("global_load_async_to_lds_b128 %0, %1, off"
                 :: "v"(lds_off), "v"(ga) : "memory");
}
__device__ __forceinline__ void wait_async0() {
    asm volatile("s_wait_asynccnt 0x0" ::: "memory");
}

// ---------------------------------------------------------------------------
// f32 -> bf16 bulk convert (8 elements / thread)
// ---------------------------------------------------------------------------
__global__ __launch_bounds__(256) void cvt_f32_bf16(
    const float* __restrict__ in, bf16_t* __restrict__ out)
{
    const size_t i = ((size_t)blockIdx.x * 256 + threadIdx.x) * 8;
    f32x4 a = *(const f32x4*)(in + i);
    f32x4 b = *(const f32x4*)(in + i + 4);
    Pack8 p;
#pragma unroll
    for (int j = 0; j < 4; ++j) { p.e[j] = (bf16_t)a[j]; p.e[4 + j] = (bf16_t)b[j]; }
    *(u32x4*)(out + i) = p.q;
}

// ---------------------------------------------------------------------------
// GEMM: C[m][n] = scale * sum_k A[m][k] * W[n][k]; A,W bf16; K = N = 1024.
// Double-buffered async-to-LDS staging; 128x128 block, 8 waves x (64x32).
// MODE 0: scatter bf16 into per-head [b][h][s][d]  MODE 1: f32 [M][1024]
// ---------------------------------------------------------------------------
#define LDT  40
#define TBUF (128 * LDT)

template<int MODE>
__global__ __launch_bounds__(256) void gemm_bf16(
    const bf16_t* __restrict__ A, const bf16_t* __restrict__ W,
    void* __restrict__ dst, float scale)
{
    __shared__ __align__(16) bf16_t As[2 * TBUF];
    __shared__ __align__(16) bf16_t Bs[2 * TBUF];

    const int tid  = threadIdx.x;
    const int wave = tid >> 5;
    const int lane = tid & 31;
    const int h    = lane >> 4;
    const int lr   = lane & 15;
    const int m0   = blockIdx.x * 128;
    const int n0   = blockIdx.y * 128;
    const int wm   = wave & 1;
    const int wn   = wave >> 1;

    auto stage = [&](int k0, int buf) {
#pragma unroll
        for (int i = 0; i < 2; ++i) {
            const int c   = tid * 2 + i;
            const int row = c >> 2;
            const int col = (c & 3) * 8;
            async_copy_b128(&As[buf * TBUF + row * LDT + col],
                            A + (size_t)(m0 + row) * D_MODEL + k0 + col);
            async_copy_b128(&Bs[buf * TBUF + row * LDT + col],
                            W + (size_t)(n0 + row) * D_MODEL + k0 + col);
        }
    };

    v8f acc[4][2] = {};
    stage(0, 0);

    for (int k0 = 0; k0 < D_MODEL; k0 += 32) {
        const int buf = (k0 >> 5) & 1;
        wait_async0();
        __syncthreads();
        if (k0 + 32 < D_MODEL) stage(k0 + 32, buf ^ 1);

        const bf16_t* Ab = &As[buf * TBUF];
        const bf16_t* Bb = &Bs[buf * TBUF];
        Frag af[4], fb[2];
#pragma unroll
        for (int mt = 0; mt < 4; ++mt) {
            const bf16_t* p = Ab + (wm * 64 + mt * 16 + lr) * LDT;
            af[mt].q[0] = *(const u32x4*)(p + 8 * h);
            af[mt].q[1] = *(const u32x4*)(p + 16 + 8 * h);
        }
#pragma unroll
        for (int nt = 0; nt < 2; ++nt) {
            const bf16_t* p = Bb + (wn * 32 + nt * 16 + lr) * LDT + 8 * h;
            fb[nt].q[0] = *(const u32x4*)p;
            fb[nt].q[1] = *(const u32x4*)(p + 16);
        }
#pragma unroll
        for (int mt = 0; mt < 4; ++mt)
#pragma unroll
            for (int nt = 0; nt < 2; ++nt)
                acc[mt][nt] = __builtin_amdgcn_wmma_f32_16x16x32_bf16(
                    false, af[mt].v, false, fb[nt].v, (short)0, acc[mt][nt], false, false);
        __syncthreads();
    }

#pragma unroll
    for (int mt = 0; mt < 4; ++mt)
#pragma unroll
        for (int nt = 0; nt < 2; ++nt) {
            const int n     = n0 + wn * 32 + nt * 16 + lr;
            const int mbase = m0 + wm * 64 + mt * 16 + 8 * h;
            if constexpr (MODE == 0) {
                const int head = n >> 6, d = n & 63;
                const int b = mbase >> 11, sbase = mbase & 2047;
                bf16_t* base = (bf16_t*)dst +
                    (((size_t)(b * NUM_HEADS + head)) * SEQ + sbase) * HEAD_DIM + d;
#pragma unroll
                for (int v = 0; v < 8; ++v)
                    base[(size_t)v * HEAD_DIM] = (bf16_t)(acc[mt][nt][v] * scale);
            } else {
                float* base = (float*)dst + (size_t)mbase * D_MODEL + n;
#pragma unroll
                for (int v = 0; v < 8; ++v)
                    base[(size_t)v * D_MODEL] = acc[mt][nt][v] * scale;
            }
        }
}

// ---------------------------------------------------------------------------
// Causal flash attention. Q is pre-scaled by 1/sqrt(HEAD_DIM) at projection.
// Q,K,V bf16 [b*H+h][s][64] -> attn bf16 [b][s][1024].
// ---------------------------------------------------------------------------
__global__ __launch_bounds__(256) void flash_attn(
    const bf16_t* __restrict__ Q, const bf16_t* __restrict__ Km,
    const bf16_t* __restrict__ V, bf16_t* __restrict__ O)
{
    __shared__ __align__(16) bf16_t Ks[128 * 72];    // [key][d]
    __shared__ __align__(16) bf16_t Vt[64 * 136];    // [d][key]
    __shared__ __align__(16) bf16_t Ps[8][16 * 136]; // per-wave P tile

    const int tid    = threadIdx.x;
    const int wave   = tid >> 5;
    const int lane   = tid & 31;
    const int h      = lane >> 4;
    const int lr     = lane & 15;
    const int qblock = blockIdx.x;
    const int bh     = blockIdx.y;
    const int q0     = qblock * 128 + wave * 16;
    const size_t seqBase = (size_t)bh * SEQ;

    Frag qa[2];
#pragma unroll
    for (int ks = 0; ks < 2; ++ks) {
        const bf16_t* qrow = Q + (seqBase + q0 + lr) * HEAD_DIM + ks * 32;
        qa[ks].q[0] = *(const u32x4*)(qrow + 8 * h);
        qa[ks].q[1] = *(const u32x4*)(qrow + 16 + 8 * h);
    }

    float mrow[8], lsum[8];
#pragma unroll
    for (int v = 0; v < 8; ++v) { mrow[v] = -3.0e38f; lsum[v] = 0.0f; }
    v8f o[4] = {};

    const int nkt = qblock + 1;
    for (int kt = 0; kt < nkt; ++kt) {
        const int kb0 = kt * 128;
        const bool needMask = (kb0 + 127) > q0;   // wave-uniform: only diagonal tile

        // ---- K tile: async global->LDS ----
#pragma unroll
        for (int i = 0; i < 4; ++i) {
            const int c   = tid + 256 * i;
            const int row = c >> 3;
            const int col = (c & 7) * 8;
            async_copy_b128(&Ks[row * 72 + col],
                            Km + (seqBase + kb0 + row) * HEAD_DIM + col);
        }
        // ---- V tile: transpose in flight -> Vt[d][key] ----
#pragma unroll
        for (int i = 0; i < 2; ++i) {
            const int pc  = tid * 2 + i;
            const int rp  = pc >> 3;
            const int col = (pc & 7) * 8;
            u32x4 r0 = *(const u32x4*)(V + (seqBase + kb0 + 2 * rp) * HEAD_DIM + col);
            u32x4 r1 = *(const u32x4*)(V + (seqBase + kb0 + 2 * rp + 1) * HEAD_DIM + col);
            const unsigned short* s0 = (const unsigned short*)&r0;
            const unsigned short* s1 = (const unsigned short*)&r1;
#pragma unroll
            for (int j = 0; j < 8; ++j) {
                const unsigned int pk = (unsigned int)s0[j] | ((unsigned int)s1[j] << 16);
                *(unsigned int*)&Vt[(col + j) * 136 + 2 * rp] = pk;
            }
        }
        wait_async0();
        __syncthreads();

        // ---- S = Q K^T (Q pre-scaled) ----
        v8f sa[8] = {};
#pragma unroll
        for (int nt = 0; nt < 8; ++nt)
#pragma unroll
            for (int ks = 0; ks < 2; ++ks) {
                Frag b;
                const bf16_t* kr = &Ks[(nt * 16 + lr) * 72 + ks * 32 + 8 * h];
                b.q[0] = *(const u32x4*)kr;
                b.q[1] = *(const u32x4*)(kr + 16);
                sa[nt] = __builtin_amdgcn_wmma_f32_16x16x32_bf16(
                    false, qa[ks].v, false, b.v, (short)0, sa[nt], false, false);
            }

        // ---- online softmax, in place on sa ----
#pragma unroll
        for (int v = 0; v < 8; ++v) {
            const int qrow = q0 + v + 8 * h;
            float mx = -3.0e38f;
            if (needMask) {
#pragma unroll
                for (int nt = 0; nt < 8; ++nt) {
                    float s = sa[nt][v];
                    const int key = kb0 + nt * 16 + lr;
                    if (key > qrow) s += -1.0e9f;    // additive causal mask
                    sa[nt][v] = s;
                    mx = fmaxf(mx, s);
                }
            } else {
#pragma unroll
                for (int nt = 0; nt < 8; ++nt) mx = fmaxf(mx, sa[nt][v]);
            }
#pragma unroll
            for (int off = 1; off < 16; off <<= 1)
                mx = fmaxf(mx, __shfl_xor(mx, off, 32));
            const float mnew  = fmaxf(mrow[v], mx);
            const float alpha = __expf(mrow[v] - mnew);
            mrow[v] = mnew;
            float rs = 0.0f;
#pragma unroll
            for (int nt = 0; nt < 8; ++nt) {
                const float p = __expf(sa[nt][v] - mnew);
                sa[nt][v] = p;
                rs += p;
            }
#pragma unroll
            for (int off = 1; off < 16; off <<= 1)
                rs += __shfl_xor(rs, off, 32);
            lsum[v] = lsum[v] * alpha + rs;
#pragma unroll
            for (int dt = 0; dt < 4; ++dt) o[dt][v] *= alpha;
        }

        // ---- P -> per-wave LDS (C layout -> A layout) ----
        bf16_t* pw = Ps[wave];
#pragma unroll
        for (int nt = 0; nt < 8; ++nt)
#pragma unroll
            for (int v = 0; v < 8; ++v)
                pw[(v + 8 * h) * 136 + nt * 16 + lr] = (bf16_t)sa[nt][v];

        // ---- O += P V ----
#pragma unroll
        for (int ks = 0; ks < 4; ++ks) {
            Frag a;
            const bf16_t* pr = &pw[lr * 136 + ks * 32];
            a.q[0] = *(const u32x4*)(pr + 8 * h);
            a.q[1] = *(const u32x4*)(pr + 16 + 8 * h);
#pragma unroll
            for (int dt = 0; dt < 4; ++dt) {
                Frag b;
                const bf16_t* vr = &Vt[(dt * 16 + lr) * 136 + ks * 32 + 8 * h];
                b.q[0] = *(const u32x4*)vr;
                b.q[1] = *(const u32x4*)(vr + 16);
                o[dt] = __builtin_amdgcn_wmma_f32_16x16x32_bf16(
                    false, a.v, false, b.v, (short)0, o[dt], false, false);
            }
        }
        __syncthreads();
    }

    // ---- normalize + write attn (bf16, [b][s][H*64]) ----
    float rinv[8];
#pragma unroll
    for (int v = 0; v < 8; ++v) rinv[v] = 1.0f / lsum[v];
    const int b = bh >> 4, head = bh & 15;
#pragma unroll
    for (int dt = 0; dt < 4; ++dt) {
        const int d = dt * 16 + lr;
        bf16_t* base = O + ((size_t)(b * SEQ + q0 + 8 * h)) * D_MODEL + head * 64 + d;
#pragma unroll
        for (int v = 0; v < 8; ++v)
            base[(size_t)v * D_MODEL] = (bf16_t)(o[dt][v] * rinv[v]);
    }
}

// ---------------------------------------------------------------------------
extern "C" void kernel_launch(void* const* d_in, const int* in_sizes, int n_in,
                              void* d_out, int out_size, void* d_ws, size_t ws_size,
                              hipStream_t stream) {
    const float* x  = (const float*)d_in[0];
    const float* Wq = (const float*)d_in[1];
    const float* Wk = (const float*)d_in[2];
    const float* Wv = (const float*)d_in[3];
    const float* Wo = (const float*)d_in[4];
    float* out = (float*)d_out;

    char* ws = (char*)d_ws;
    const size_t XB  = (size_t)M_TOT * D_MODEL * sizeof(bf16_t);    // 16 MB
    const size_t WB  = (size_t)D_MODEL * D_MODEL * sizeof(bf16_t);  //  2 MB
    bf16_t* xb   = (bf16_t*)(ws);
    bf16_t* wqb  = (bf16_t*)(ws + XB);
    bf16_t* wkb  = (bf16_t*)(ws + XB + WB);
    bf16_t* wvb  = (bf16_t*)(ws + XB + 2 * WB);
    bf16_t* wob  = (bf16_t*)(ws + XB + 3 * WB);
    bf16_t* qb   = (bf16_t*)(ws + XB + 4 * WB);
    bf16_t* kb   = (bf16_t*)(ws + 2 * XB + 4 * WB);
    bf16_t* vb   = (bf16_t*)(ws + 3 * XB + 4 * WB);
    bf16_t* attn = (bf16_t*)(ws + 4 * XB + 4 * WB);                 // 88 MB total

    dim3 blk(256);

    cvt_f32_bf16<<<dim3(M_TOT * D_MODEL / 2048), blk, 0, stream>>>(x,  xb);
    cvt_f32_bf16<<<dim3(D_MODEL * D_MODEL / 2048), blk, 0, stream>>>(Wq, wqb);
    cvt_f32_bf16<<<dim3(D_MODEL * D_MODEL / 2048), blk, 0, stream>>>(Wk, wkb);
    cvt_f32_bf16<<<dim3(D_MODEL * D_MODEL / 2048), blk, 0, stream>>>(Wv, wvb);
    cvt_f32_bf16<<<dim3(D_MODEL * D_MODEL / 2048), blk, 0, stream>>>(Wo, wob);

    dim3 gemmGrid(M_TOT / 128, D_MODEL / 128);
    // Q pre-scaled by 1/sqrt(HEAD_DIM) = 0.125 (exact in bf16)
    gemm_bf16<0><<<gemmGrid, blk, 0, stream>>>(xb, wqb, (void*)qb, 0.125f);
    gemm_bf16<0><<<gemmGrid, blk, 0, stream>>>(xb, wkb, (void*)kb, 1.0f);
    gemm_bf16<0><<<gemmGrid, blk, 0, stream>>>(xb, wvb, (void*)vb, 1.0f);

    dim3 faGrid(SEQ / 128, BATCH * NUM_HEADS);
    flash_attn<<<faGrid, blk, 0, stream>>>(qb, kb, vb, attn);

    gemm_bf16<1><<<gemmGrid, blk, 0, stream>>>(attn, wob, (void*)out, 1.0f);
}
